// BiLSTMDiacritizer_49074296324800
// MI455X (gfx1250) — matrix operand order repaired and checked by reading, not compiled, not run on hardware
//
#include <hip/hip_runtime.h>
#include <hip/hip_bf16.h>

// ---------------------------------------------------------------------------
// BiLSTM diacritizer forward, CDNA5 (gfx1250), wave32, WMMA f16->f32,
// TDM-staged GEMM A-tiles, asm-streamed recurrent weights.
// ---------------------------------------------------------------------------
#define VSZ 64
#define EMB 128
#define HH 256
#define CLS 15
#define H2 512
#define BB 4
#define SS 256
#define BS (BB * SS)      // 1024 rows
#define G4H (4 * HH)      // 1024 gate columns

typedef __attribute__((ext_vector_type(16))) _Float16     v16h;
typedef __attribute__((ext_vector_type(8)))  float        v8f;
typedef __attribute__((ext_vector_type(4)))  unsigned int v4u_t;
typedef __attribute__((ext_vector_type(8)))  int          v8i_t;
typedef __attribute__((ext_vector_type(4)))  int          v4i_t;

union Frag {
    v16h v;
    unsigned int u[8];
    _Float16 h[16];
};

// ---- WMMA fragment loaders (contiguous k order, same in A and B) -----------------
// A: 16x32 f16 (MxK). lane = m (0..15) in each half; VGPR j holds K pairs:
//   K = 16*half + 2j, 2j+1 (j = 0..7)
__device__ inline void load_a_frag(const _Float16* __restrict__ A, int lda,
                                   int row0, int k0, int lane, Frag& a) {
    const int m = lane & 15, half = lane >> 4;
    const _Float16* base = A + (size_t)(row0 + m) * lda + k0 + 16 * half;
#pragma unroll
    for (int j = 0; j < 4; ++j)
        a.u[j] = *reinterpret_cast<const unsigned int*>(base + 2 * j);
#pragma unroll
    for (int j = 0; j < 4; ++j)
        a.u[4 + j] = *reinterpret_cast<const unsigned int*>(base + 8 + 2 * j);
}

// B: 32x16 f16 (KxN), source = weights W (N x K) row-major (K contiguous).
// lanes 0-15: K 0..15 (pairs in VGPRs 0..7); lanes 16-31: K 16..31.
__device__ inline void load_b_frag_wt(const _Float16* __restrict__ W, int ldw,
                                      int ncol0, int k0, int lane, Frag& b) {
    const int n = ncol0 + (lane & 15);
    const int kb = k0 + 16 * (lane >> 4);
    const _Float16* base = W + (size_t)n * ldw + kb;
#pragma unroll
    for (int j = 0; j < 8; ++j)
        b.u[j] = *reinterpret_cast<const unsigned int*>(base + 2 * j);
}

// B source stored as (K x N) row-major: K-pairs strided, pack 16-bit loads.
__device__ inline void load_b_frag_kn(const _Float16* __restrict__ Bm, int ldb,
                                      int ncol0, int k0, int lane, Frag& b) {
    const int n = ncol0 + (lane & 15);
    const int kb = k0 + 16 * (lane >> 4);
#pragma unroll
    for (int j = 0; j < 8; ++j) {
        b.h[2 * j]     = Bm[(size_t)(kb + 2 * j) * ldb + n];
        b.h[2 * j + 1] = Bm[(size_t)(kb + 2 * j + 1) * ldb + n];
    }
}

// Streaming (never hoisted, never spilled) load of TWO adjacent B tiles
// (32 cols) for W_hh with ldw==HH: 4x global_load_b128 + explicit
// s_wait_loadcnt in one asm blob. Tile1 rows are +16*HH halves = 8192 B away.
__device__ inline void load_b2_stream_hh(const _Float16* W, int ncol0, int k0,
                                         int lane, Frag& b0, Frag& b1) {
    const int n = ncol0 + (lane & 15);
    const int kb = k0 + 16 * (lane >> 4);
    const _Float16* base = W + (size_t)n * HH + kb;
    v4u_t a0, a1, c0, c1;
    asm volatile(
        "global_load_b128 %0, %4, off\n\t"
        "global_load_b128 %1, %4, off offset:16\n\t"
        "global_load_b128 %2, %4, off offset:8192\n\t"
        "global_load_b128 %3, %4, off offset:8208\n\t"
        "s_wait_loadcnt 0x0"
        : "=v"(a0), "=v"(a1), "=v"(c0), "=v"(c1)
        : "v"((unsigned long long)(const void*)base)
        : "memory");
#pragma unroll
    for (int j = 0; j < 4; ++j) {
        b0.u[j] = a0[j]; b0.u[4 + j] = a1[j];
        b1.u[j] = c0[j]; b1.u[4 + j] = c1[j];
    }
}

// ---------------------------------------------------------------------------
// Tiled WMMA GEMM:  D(M,N) f32 = A(M,K)f16 * op(B) + bias[n]
// BT=0: B = weights (N x K) row-major.  BT=1: B = (K x N) row-major.
// Block: 128 threads = 4 waves; wave owns 2 adjacent 16-col tiles.
// grid = (M/16, N/128, batch).  Requires lda == K (true for every call here)
// and K in {128,256,512}: the 16xK A tile is staged into LDS by the Tensor
// Data Mover with +4dword/row padding (bank-conflict-free A-frag reads).
// ---------------------------------------------------------------------------
#define APADH 8  // LDS A row padding in halves (4 dwords)

template <int BT>
__global__ __launch_bounds__(128)
void wmma_gemm_kernel(const _Float16* __restrict__ A, int lda, long strideA,
                      const _Float16* __restrict__ Bm, int ldb, long strideB,
                      const float* __restrict__ bias,
                      float* __restrict__ D, int ldd, long strideD, int K) {
    __shared__ _Float16 As[16 * (H2 + APADH)];  // max K=512
    const int tid = threadIdx.x;
    const int wave = tid >> 5, lane = tid & 31;
    const int row0 = blockIdx.x * 16;
    const int ncol0 = blockIdx.y * 128 + wave * 32;
    const _Float16* Ab = A + (size_t)blockIdx.z * strideA;
    const _Float16* Bb = Bm + (size_t)blockIdx.z * strideB;
    float* Db = D + (size_t)blockIdx.z * strideD;

    // ---- TDM: stage 16 x K f16 A-tile into LDS (wave 0 issues) ------------
    if (tid < 32) {
        const unsigned lds_off = (unsigned)(size_t)(&As[0]);
        const unsigned long long ga =
            (unsigned long long)(const void*)(Ab + (size_t)row0 * lda);
        const int pi = __builtin_ctz((unsigned)K) - 2;  // row = (2<<pi) dwords
        v4u_t g0;
        g0[0] = 1u;                                   // count=1, user desc
        g0[1] = lds_off;                              // lds_addr (bytes)
        g0[2] = (unsigned)(ga & 0xFFFFFFFFu);         // global_addr[31:0]
        g0[3] = (unsigned)((ga >> 32) & 0x1FFFFFFu)   // global_addr[56:32]
                | (2u << 30);                         // type=2 ("image")
        v8i_t g1 = 0;
        g1[0] = (1 << 16)                             // data_size=1 -> 2 bytes
                | (1 << 20)                           // pad_enable
                | (pi << 22)                          // pad_interval: row dwords
                | (3 << 25);                          // pad_amount: 4 dwords
        g1[1] = (K & 0xFFFF) << 16;                   // tensor_dim0[15:0]
        g1[2] = 16 << 16;                             // tensor_dim1[15:0] = 16
        g1[3] = (K & 0xFFFF) << 16;                   // tile_dim0 = K
        g1[4] = 16;                                   // tile_dim1 = 16 rows
        g1[5] = lda;                                  // tensor_dim0_stride
        v4i_t gz = 0;
#if __clang_major__ >= 23
        v8i_t gz8 = 0;
        __builtin_amdgcn_tensor_load_to_lds(g0, g1, gz, gz, gz8, 0);
#else
        __builtin_amdgcn_tensor_load_to_lds(g0, g1, gz, gz, 0);
#endif
        __builtin_amdgcn_s_wait_tensorcnt((short)0);
    }
    __syncthreads();
    const int ldaS = K + APADH;  // padded LDS row stride (halves)

    v8f acc0 = {};
    v8f acc1 = {};
    for (int k0 = 0; k0 < K; k0 += 32) {
        if (k0 + 32 < K)  // prefetch next K-slab of B (global_prefetch_b8)
            __builtin_prefetch(Bb + (size_t)ncol0 * ldb + (k0 + 32), 0, 1);
        Frag a, b0, b1;
        load_a_frag(As, ldaS, 0, k0, lane, a);
        if (BT == 0) {
            load_b_frag_wt(Bb, ldb, ncol0, k0, lane, b0);
            load_b_frag_wt(Bb, ldb, ncol0 + 16, k0, lane, b1);
        } else {
            load_b_frag_kn(Bb, ldb, ncol0, k0, lane, b0);
            load_b_frag_kn(Bb, ldb, ncol0 + 16, k0, lane, b1);
        }
        acc0 = __builtin_amdgcn_wmma_f32_16x16x32_f16(false, a.v, false, b0.v,
                                                      (short)0, acc0, false, false);
        acc1 = __builtin_amdgcn_wmma_f32_16x16x32_f16(false, a.v, false, b1.v,
                                                      (short)0, acc1, false, false);
    }
    const int colB = ncol0 + (lane & 15);
    const int rowB = row0 + 8 * (lane >> 4);
    float bc0 = 0.f, bc1 = 0.f;
    if (bias) { bc0 = bias[colB]; bc1 = bias[colB + 16]; }
#pragma unroll
    for (int r = 0; r < 8; ++r) {
        Db[(size_t)(rowB + r) * ldd + colB]      = acc0[r] + bc0;
        Db[(size_t)(rowB + r) * ldd + colB + 16] = acc1[r] + bc1;
    }
}

// ---------------------------------------------------------------------------
// Sequential LSTM over time, one workgroup per direction (grid.x = 2).
// 1024 threads = 32 waves; h kept in LDS as padded f16 16xHH tile; W_hh f16
// streamed from L2 every step via asm global_load_b128 (no spills).
// ---------------------------------------------------------------------------
#define HPAD 264  // padded LDS row stride in halves

__device__ inline float sigmoidf_(float x) { return 1.0f / (1.0f + expf(-x)); }

__global__ __launch_bounds__(1024)
void lstm_seq_kernel(const float* __restrict__ gx, long gxDirStride,
                     const _Float16* __restrict__ whh, long whhDirStride,
                     _Float16* __restrict__ hs_h /* (B*S, H2) f16 */) {
    __shared__ _Float16 h_sh[16 * HPAD];
    __shared__ float g_s[4 * G4H];

    const int tid = threadIdx.x;
    const int dir = blockIdx.x;
    const int wave = tid >> 5, lane = tid & 31;
    const int ncb = wave * 32;
    const int b_ = tid >> 8;   // 0..3
    const int n_ = tid & 255;  // 0..255
    const int colOff = dir * HH;

    const float* gxd = gx + (size_t)dir * gxDirStride;
    const _Float16* whd = whh + (size_t)dir * whhDirStride;

    for (int i = tid; i < 16 * HPAD; i += 1024) h_sh[i] = (_Float16)0.0f;
    float c_reg = 0.0f;
    __syncthreads();

    for (int tt = 0; tt < SS; ++tt) {
        const int t = dir ? (SS - 1 - tt) : tt;

        // ---- recurrent matmul: g = h_pad(16x256) * Whh^T slice (32 cols) ---
        v8f acc0 = {};
        v8f acc1 = {};
#pragma unroll
        for (int kt = 0; kt < 8; ++kt) {
            const int k0 = kt * 32;
            Frag a, b0, b1;
            load_a_frag(h_sh, HPAD, 0, k0, lane, a);
            load_b2_stream_hh(whd, ncb, k0, lane, b0, b1);
            acc0 = __builtin_amdgcn_wmma_f32_16x16x32_f16(false, a.v, false, b0.v,
                                                          (short)0, acc0, false, false);
            acc1 = __builtin_amdgcn_wmma_f32_16x16x32_f16(false, a.v, false, b1.v,
                                                          (short)0, acc1, false, false);
        }
        // rows 0..3 (real batch rows) live in VGPRs 0..3 of lanes 0..15
        if (lane < 16) {
#pragma unroll
            for (int r = 0; r < 4; ++r) {
                g_s[r * G4H + ncb + lane]      = acc0[r];
                g_s[r * G4H + ncb + 16 + lane] = acc1[r];
            }
        }
        __syncthreads();

        // ---- pointwise gates; each thread owns one (b, n) -----------------
        const float* gxr = gxd + ((size_t)(b_ * SS + t)) * G4H;
        const float gi = g_s[b_ * G4H + n_]       + gxr[n_];
        const float gf = g_s[b_ * G4H + 256 + n_] + gxr[256 + n_];
        const float gg = g_s[b_ * G4H + 512 + n_] + gxr[512 + n_];
        const float go = g_s[b_ * G4H + 768 + n_] + gxr[768 + n_];
        const float iv = sigmoidf_(gi);
        const float fv = sigmoidf_(gf);
        const float gv = tanhf(gg);
        const float ov = sigmoidf_(go);
        c_reg = fv * c_reg + iv * gv;
        const float hv = ov * tanhf(c_reg);

        h_sh[b_ * HPAD + n_] = (_Float16)hv;
        hs_h[((size_t)(b_ * SS + t)) * H2 + colOff + n_] = (_Float16)hv;
        __syncthreads();
    }
}

// ---------------------------------------------------------------------------
// Small kernels
// ---------------------------------------------------------------------------
__global__ void cvt_f32_to_f16_kernel(const float* __restrict__ src,
                                      _Float16* __restrict__ dst, int n) {
    int i = blockIdx.x * blockDim.x + threadIdx.x;
    if (i < n) dst[i] = (_Float16)src[i];
}

__global__ void embed_kernel(const int* __restrict__ ids,
                             const float* __restrict__ emb,
                             _Float16* __restrict__ xh) {
    const int r = blockIdx.x;   // b*S + s
    const int e = threadIdx.x;  // 0..127
    const int id = ids[r];
    xh[(size_t)r * EMB + e] = (_Float16)emb[(size_t)id * EMB + e];
}

// scores[b,t,s] = sum_h tanh(q[b,t,h] + k[b,s,h]) * v[h]
__global__ __launch_bounds__(256)
void attn_scores_kernel(const float* __restrict__ q, const float* __restrict__ k,
                        const float* __restrict__ v, float* __restrict__ scores) {
    __shared__ float qs[H2];
    __shared__ float vs[H2];
    const int bt = blockIdx.x;  // b*S + t
    const int b = bt >> 8;
    const int tid = threadIdx.x;
    for (int i = tid; i < H2; i += 256) {
        qs[i] = q[(size_t)bt * H2 + i];
        vs[i] = v[i];
    }
    __syncthreads();
    const int s = tid;
    const float* kr = k + ((size_t)(b * SS + s)) * H2;
    float acc = 0.0f;
#pragma unroll 4
    for (int h = 0; h < H2; ++h) acc += tanhf(qs[h] + kr[h]) * vs[h];
    scores[(size_t)bt * SS + s] = acc;
}

__global__ __launch_bounds__(256)
void softmax_kernel(const float* __restrict__ scores, _Float16* __restrict__ w) {
    __shared__ float red[256];
    const int bt = blockIdx.x;
    const int tid = threadIdx.x;
    const float x = scores[(size_t)bt * SS + tid];
    red[tid] = x;
    __syncthreads();
    for (int off = 128; off > 0; off >>= 1) {
        if (tid < off) red[tid] = fmaxf(red[tid], red[tid + off]);
        __syncthreads();
    }
    const float mx = red[0];
    __syncthreads();
    const float e = expf(x - mx);
    red[tid] = e;
    __syncthreads();
    for (int off = 128; off > 0; off >>= 1) {
        if (tid < off) red[tid] += red[tid + off];
        __syncthreads();
    }
    w[(size_t)bt * SS + tid] = (_Float16)(e / red[0]);
}

// out[b,t,c] = ctx[b,t,:] . cls_W[c,:] + cls_b[c]   (one wave per output)
__global__ __launch_bounds__(32)
void logits_kernel(const float* __restrict__ ctx, const float* __restrict__ W,
                   const float* __restrict__ bias, float* __restrict__ out) {
    const int bt = blockIdx.x;
    const int c = blockIdx.y;
    const int lane = threadIdx.x;
    const float* cr = ctx + (size_t)bt * H2;
    const float* wr = W + (size_t)c * H2;
    float acc = 0.0f;
    for (int h = lane; h < H2; h += 32) acc += cr[h] * wr[h];
#pragma unroll
    for (int off = 16; off > 0; off >>= 1) acc += __shfl_down(acc, off, 32);
    if (lane == 0) out[(size_t)bt * CLS + c] = acc + bias[c];
}

// ---------------------------------------------------------------------------
// Host launch
// ---------------------------------------------------------------------------
static inline char* wsAlloc(char*& p, size_t bytes) {
    char* r = p;
    p += (bytes + 255) & ~(size_t)255;
    return r;
}

extern "C" void kernel_launch(void* const* d_in, const int* in_sizes, int n_in,
                              void* d_out, int out_size, void* d_ws, size_t ws_size,
                              hipStream_t stream) {
    const int*   ids    = (const int*)d_in[0];
    const float* emb    = (const float*)d_in[1];
    const float* w_ih0  = (const float*)d_in[2];   // (2,1024,128)
    const float* w_hh0  = (const float*)d_in[3];   // (2,1024,256)
    const float* b0     = (const float*)d_in[4];   // (2,1024)
    const float* w_ih   = (const float*)d_in[5];   // (2,2,1024,512)
    const float* w_hh   = (const float*)d_in[6];   // (2,2,1024,256)
    const float* bb     = (const float*)d_in[7];   // (2,2,1024)
    const float* attn_W = (const float*)d_in[8];   // (512,512)
    const float* attn_U = (const float*)d_in[9];   // (512,512)
    const float* attn_v = (const float*)d_in[10];  // (512,)
    const float* cls_W  = (const float*)d_in[11];  // (15,512)
    const float* cls_b  = (const float*)d_in[12];  // (15,)
    float* out = (float*)d_out;

    char* p = (char*)d_ws;
    _Float16* xh     = (_Float16*)wsAlloc(p, (size_t)BS * EMB * 2);
    _Float16* wih0h  = (_Float16*)wsAlloc(p, (size_t)2 * G4H * EMB * 2);
    _Float16* whh0h  = (_Float16*)wsAlloc(p, (size_t)2 * G4H * HH * 2);
    _Float16* wihh   = (_Float16*)wsAlloc(p, (size_t)4 * G4H * H2 * 2);
    _Float16* whhh   = (_Float16*)wsAlloc(p, (size_t)4 * G4H * HH * 2);
    _Float16* attnWh = (_Float16*)wsAlloc(p, (size_t)H2 * H2 * 2);
    _Float16* attnUh = (_Float16*)wsAlloc(p, (size_t)H2 * H2 * 2);
    float*    gx     = (float*)wsAlloc(p, (size_t)2 * BS * G4H * 4);
    _Float16* hA     = (_Float16*)wsAlloc(p, (size_t)BS * H2 * 2);
    _Float16* hB     = (_Float16*)wsAlloc(p, (size_t)BS * H2 * 2);
    float*    qbuf   = (float*)wsAlloc(p, (size_t)BS * H2 * 4);
    float*    kbuf   = (float*)wsAlloc(p, (size_t)BS * H2 * 4);
    float*    sc     = (float*)wsAlloc(p, (size_t)BS * SS * 4);
    _Float16* wsoft  = (_Float16*)wsAlloc(p, (size_t)BS * SS * 2);
    float*    ctx    = (float*)wsAlloc(p, (size_t)BS * H2 * 4);

    auto cvt = [&](const float* s, _Float16* d, int n) {
        cvt_f32_to_f16_kernel<<<(n + 255) / 256, 256, 0, stream>>>(s, d, n);
    };
    cvt(w_ih0, wih0h, 2 * G4H * EMB);
    cvt(w_hh0, whh0h, 2 * G4H * HH);
    cvt(w_ih, wihh, 4 * G4H * H2);
    cvt(w_hh, whhh, 4 * G4H * HH);
    cvt(attn_W, attnWh, H2 * H2);
    cvt(attn_U, attnUh, H2 * H2);

    embed_kernel<<<BS, EMB, 0, stream>>>(ids, emb, xh);

    // ---- layer 0 -----------------------------------------------------------
    for (int d = 0; d < 2; ++d) {
        wmma_gemm_kernel<0><<<dim3(BS / 16, G4H / 128, 1), 128, 0, stream>>>(
            xh, EMB, 0, wih0h + (size_t)d * G4H * EMB, EMB, 0,
            b0 + (size_t)d * G4H, gx + (size_t)d * BS * G4H, G4H, 0, EMB);
    }
    lstm_seq_kernel<<<2, 1024, 0, stream>>>(gx, (long)BS * G4H,
                                            whh0h, (long)G4H * HH, hA);

    // ---- layers 1 and 2 ----------------------------------------------------
    for (int l = 0; l < 2; ++l) {
        _Float16* hin = (l == 0) ? hA : hB;
        _Float16* hout = (l == 0) ? hB : hA;
        for (int d = 0; d < 2; ++d) {
            const int w = l * 2 + d;
            wmma_gemm_kernel<0><<<dim3(BS / 16, G4H / 128, 1), 128, 0, stream>>>(
                hin, H2, 0, wihh + (size_t)w * G4H * H2, H2, 0,
                bb + (size_t)w * G4H, gx + (size_t)d * BS * G4H, G4H, 0, H2);
        }
        lstm_seq_kernel<<<2, 1024, 0, stream>>>(gx, (long)BS * G4H,
                                                whhh + (size_t)l * 2 * G4H * HH,
                                                (long)G4H * HH, hout);
    }
    _Float16* hfin = hA;  // final layer output

    // ---- attention projections q, k ---------------------------------------
    wmma_gemm_kernel<0><<<dim3(BS / 16, H2 / 128, 1), 128, 0, stream>>>(
        hfin, H2, 0, attnWh, H2, 0, nullptr, qbuf, H2, 0, H2);
    wmma_gemm_kernel<0><<<dim3(BS / 16, H2 / 128, 1), 128, 0, stream>>>(
        hfin, H2, 0, attnUh, H2, 0, nullptr, kbuf, H2, 0, H2);

    // ---- additive attention scores + softmax ------------------------------
    attn_scores_kernel<<<BS, 256, 0, stream>>>(qbuf, kbuf, attn_v, sc);
    softmax_kernel<<<BS, 256, 0, stream>>>(sc, wsoft);

    // ---- context: per-batch GEMM  ctx[b] = w[b](SxS) * h[b](SxH2) ---------
    wmma_gemm_kernel<1><<<dim3(SS / 16, H2 / 128, BB), 128, 0, stream>>>(
        wsoft, SS, (long)SS * SS, hfin, H2, (long)SS * H2, nullptr,
        ctx, H2, (long)SS * H2, SS);

    // ---- classifier --------------------------------------------------------
    logits_kernel<<<dim3(BS, CLS, 1), 32, 0, stream>>>(ctx, cls_W, cls_b, out);
}
